// TransformerEncoderLayer_76553497084438
// MI455X (gfx1250) — matrix operand order repaired and checked by reading, not compiled, not run on hardware
//
#include <hip/hip_runtime.h>
#include <hip/hip_bf16.h>
#include <math.h>

#define D_MODEL   512
#define NHEAD     8
#define HEAD_DIM  64
#define T_SEQ     1024
#define N_BATCH   8
#define M_ROWS    (N_BATCH * T_SEQ)      // 8192
#define H_FF      2048
#define O_QKV     (3 * D_MODEL)          // 1536
#define ALPHA_F   2.4495f
#define EPS_F     1e-5f
#define WIN_L     127
#define WIN_R     128

typedef _Float16 h16 __attribute__((ext_vector_type(16)));
typedef float    f8  __attribute__((ext_vector_type(8)));

static __device__ __forceinline__ f8 f8zero() {
  f8 z;
#pragma unroll
  for (int i = 0; i < 8; ++i) z[i] = 0.0f;
  return z;
}

static __device__ __forceinline__ f8 wmma16(h16 a, h16 b, f8 c) {
  // D = A(16x32 f16) * B(32x16 f16) + C(16x16 f32)
  return __builtin_amdgcn_wmma_f32_16x16x32_f16(false, a, false, b, (short)0, c,
                                                false, false);
}

// ---- fragment loaders (CDNA5 wave32 WMMA layouts) -------------------------
// A 16x32 f16: lane L -> row m = L%16, h = L/16; vector j -> K = 16h + j
static __device__ __forceinline__ h16 ldA_f32(const float* A, int lda, int m0,
                                              int k0, int lane) {
  const float* p = A + (size_t)(m0 + (lane & 15)) * lda + k0 + 16 * (lane >> 4);
  h16 a;
#pragma unroll
  for (int j = 0; j < 8; ++j) {
    a[j]     = (_Float16)p[j];
    a[8 + j] = (_Float16)p[8 + j];
  }
  return a;
}

static __device__ __forceinline__ h16 ldA_f16(const _Float16* A, int lda, int m0,
                                              int k0, int lane) {
  const _Float16* p =
      A + (size_t)(m0 + (lane & 15)) * lda + k0 + 16 * (lane >> 4);
  h16 a;
#pragma unroll
  for (int j = 0; j < 8; ++j) {
    a[j]     = p[j];
    a[8 + j] = p[8 + j];
  }
  return a;
}

// B 32x16 f16: lane L -> col n = L%16, h = L/16; vector j -> K = 16h + j.
// Source is row-major (O x K) weights: column n of B == row (n0+n) of W.
static __device__ __forceinline__ h16 ldB_rowmaj(const _Float16* W, int ldb,
                                                 int n0, int k0, int lane) {
  const _Float16* p =
      W + (size_t)(n0 + (lane & 15)) * ldb + k0 + 16 * (lane >> 4);
  h16 b;
#pragma unroll
  for (int j = 0; j < 16; ++j) b[j] = p[j];
  return b;
}

// ---- weight conversion -----------------------------------------------------
__global__ __launch_bounds__(256) void k_cvt(const float* __restrict__ in,
                                             _Float16* __restrict__ out, int n) {
  int i = blockIdx.x * 256 + threadIdx.x;
  if (i < n) out[i] = (_Float16)in[i];
}

// ---- QKV GEMM: qkv = x @ Wqkv^T, scatter into q/k (N,H,T,Dh) and vT (N,H,Dh,T)
// 32x64 macro-tile per wave: 2 M-tiles x 4 N-tiles
__global__ __launch_bounds__(256) void k_qkv(const float* __restrict__ x,
                                             const _Float16* __restrict__ Wh,
                                             _Float16* __restrict__ qh,
                                             _Float16* __restrict__ kh,
                                             _Float16* __restrict__ vTh) {
  int lane = threadIdx.x & 31;
  int gw   = blockIdx.x * 8 + (threadIdx.x >> 5);   // 6144 waves
  int ntg  = gw % (O_QKV / 64);                     // 24 groups of 4 n-tiles
  int mt   = gw / (O_QKV / 64);                     // 256 m-blocks of 32
  int m0 = mt * 32, n0 = ntg * 64;

  f8 acc[2][4];
#pragma unroll
  for (int mi = 0; mi < 2; ++mi)
#pragma unroll
    for (int nt = 0; nt < 4; ++nt) acc[mi][nt] = f8zero();

  for (int k0 = 0; k0 < D_MODEL; k0 += 32) {
    h16 a0 = ldA_f32(x, D_MODEL, m0, k0, lane);
    h16 a1 = ldA_f32(x, D_MODEL, m0 + 16, k0, lane);
#pragma unroll
    for (int nt = 0; nt < 4; ++nt) {
      h16 b = ldB_rowmaj(Wh, D_MODEL, n0 + 16 * nt, k0, lane);
      acc[0][nt] = wmma16(a0, b, acc[0][nt]);
      acc[1][nt] = wmma16(a1, b, acc[1][nt]);
    }
  }
  int n = lane & 15, hh = lane >> 4;
#pragma unroll
  for (int mi = 0; mi < 2; ++mi) {
#pragma unroll
    for (int nt = 0; nt < 4; ++nt) {
#pragma unroll
      for (int r = 0; r < 8; ++r) {
        int m = m0 + 16 * mi + r + 8 * hh;
        int o = n0 + nt * 16 + n;
        int b_ = m >> 10, t = m & 1023;
        int which = o >> 9, oo = o & 511;
        int hd = oo >> 6, d = oo & 63;
        _Float16 hv = (_Float16)acc[mi][nt][r];
        size_t headbase = (size_t)(b_ * NHEAD + hd);
        if (which == 0)
          qh[(headbase * T_SEQ + t) * HEAD_DIM + d] = hv;
        else if (which == 1)
          kh[(headbase * T_SEQ + t) * HEAD_DIM + d] = hv;
        else
          vTh[(headbase * HEAD_DIM + d) * T_SEQ + t] = hv;
      }
    }
  }
}

// ---- RoPE in-place on q and k ---------------------------------------------
__global__ __launch_bounds__(256) void k_rope(_Float16* __restrict__ qh,
                                              _Float16* __restrict__ kh) {
  int idx = blockIdx.x * 256 + threadIdx.x;         // N*H*T*32
  if (idx >= N_BATCH * NHEAD * T_SEQ * 32) return;
  int dp = idx & 31;
  int t  = (idx >> 5) & (T_SEQ - 1);
  int bh = idx >> 15;
  float inv = __powf(10000.0f, -(float)(2 * dp) / (float)HEAD_DIM);
  float ang = (float)t * inv;
  float c = __cosf(ang), s = __sinf(ang);
  size_t base = ((size_t)bh * T_SEQ + t) * HEAD_DIM;
  float q1 = (float)qh[base + dp], q2 = (float)qh[base + dp + 32];
  qh[base + dp]      = (_Float16)(q1 * c - q2 * s);
  qh[base + dp + 32] = (_Float16)(q2 * c + q1 * s);
  float k1 = (float)kh[base + dp], k2 = (float)kh[base + dp + 32];
  kh[base + dp]      = (_Float16)(k1 * c - k2 * s);
  kh[base + dp + 32] = (_Float16)(k2 * c + k1 * s);
}

// ---- banded flash attention: one wave per (b, h, 16-query tile) -----------
__global__ __launch_bounds__(256) void k_attn(const _Float16* __restrict__ qh,
                                              const _Float16* __restrict__ kh,
                                              const _Float16* __restrict__ vTh,
                                              _Float16* __restrict__ ctx) {
  __shared__ _Float16 Pls[8][16][40];               // per-wave P transpose buffer
  const float NEGINF = -__builtin_inff();
  int w = threadIdx.x >> 5, lane = threadIdx.x & 31;
  int gw = blockIdx.x * 8 + w;                      // 4096 waves
  int it = gw & 63;
  int h  = (gw >> 6) & (NHEAD - 1);
  int b  = gw >> 9;
  const _Float16* q  = qh  + ((size_t)(b * NHEAD + h)) * T_SEQ * HEAD_DIM;
  const _Float16* kk = kh  + ((size_t)(b * NHEAD + h)) * T_SEQ * HEAD_DIM;
  const _Float16* vT = vTh + ((size_t)(b * NHEAD + h)) * HEAD_DIM * T_SEQ;
  int i0 = it * 16;
  int n = lane & 15, hh = lane >> 4;
  const float scale = 0.125f;                       // 1/sqrt(64)

  f8 acc[4] = {f8zero(), f8zero(), f8zero(), f8zero()};
  float mrow[8], lrow[8];
#pragma unroll
  for (int r = 0; r < 8; ++r) { mrow[r] = NEGINF; lrow[r] = 0.0f; }

  for (int p = 0; p < 9; ++p) {
    int jt0 = it - 8 + 2 * p;                       // pair of key tiles
    f8 s[2] = {f8zero(), f8zero()};
#pragma unroll
    for (int kt = 0; kt < 2; ++kt) {
      int jb = (jt0 + kt) * 16;
      int jbc = jb < 0 ? 0 : (jb > T_SEQ - 16 ? T_SEQ - 16 : jb);
#pragma unroll
      for (int kc = 0; kc < HEAD_DIM; kc += 32) {
        h16 a  = ldA_f16(q, HEAD_DIM, i0, kc, lane);
        h16 bm = ldB_rowmaj(kk, HEAD_DIM, jbc, kc, lane);  // cols = keys
        s[kt]  = wmma16(a, bm, s[kt]);
      }
    }
    // mask + online softmax (rows r+8*hh live across 16-lane half groups)
    float mnew[8];
#pragma unroll
    for (int r = 0; r < 8; ++r) {
      int i = i0 + r + 8 * hh;
      float vmax = NEGINF;
#pragma unroll
      for (int kt = 0; kt < 2; ++kt) {
        int j = (jt0 + kt) * 16 + n;
        int df = j - i;
        bool ok = (j >= 0) && (j < T_SEQ) && (df >= -WIN_L) && (df <= WIN_R);
        float sv = s[kt][r] * scale;
        sv = ok ? sv : NEGINF;
        s[kt][r] = sv;
        vmax = fmaxf(vmax, sv);
      }
#pragma unroll
      for (int off = 1; off < 16; off <<= 1)
        vmax = fmaxf(vmax, __shfl_xor(vmax, off, 32));
      mnew[r] = fmaxf(mrow[r], vmax);
      float sc = (mrow[r] > NEGINF && mnew[r] > NEGINF)
                     ? __expf(mrow[r] - mnew[r]) : (mnew[r] > NEGINF ? 0.0f : 1.0f);
      float rsum = 0.0f;
#pragma unroll
      for (int kt = 0; kt < 2; ++kt) {
        float e = (s[kt][r] > NEGINF) ? __expf(s[kt][r] - mnew[r]) : 0.0f;
        s[kt][r] = e;
        rsum += e;
      }
#pragma unroll
      for (int off = 1; off < 16; off <<= 1) rsum += __shfl_xor(rsum, off, 32);
      lrow[r] = lrow[r] * sc + rsum;
#pragma unroll
      for (int dt = 0; dt < 4; ++dt) acc[dt][r] *= sc;
      mrow[r] = mnew[r];
      // stash P in LDS (C-layout -> memory row-major 16x32)
#pragma unroll
      for (int kt = 0; kt < 2; ++kt)
        Pls[w][r + 8 * hh][kt * 16 + n] = (_Float16)s[kt][r];
    }
#if __has_builtin(__builtin_amdgcn_wave_barrier)
    __builtin_amdgcn_wave_barrier();
#else
    asm volatile("" ::: "memory");
#endif
    // reload P as an A fragment (16x32)
    h16 ap;
    {
      const _Float16* pp = &Pls[w][lane & 15][16 * hh];
#pragma unroll
      for (int j = 0; j < 8; ++j) {
        ap[j]     = pp[j];
        ap[8 + j] = pp[8 + j];
      }
    }
    int jb0 = jt0 * 16;
#pragma unroll
    for (int dt = 0; dt < 4; ++dt) {
      int d = dt * 16 + n;
      int jb = jb0 + 16 * hh;
      jb = jb < 0 ? 0 : (jb > T_SEQ - 16 ? T_SEQ - 16 : jb);
      const _Float16* pv = vT + (size_t)d * T_SEQ + jb;
      h16 bv;
#pragma unroll
      for (int j = 0; j < 16; ++j) bv[j] = pv[j];
      acc[dt] = wmma16(ap, bv, acc[dt]);
    }
  }
  // normalize and store ctx in (N*T, D) f16, col = h*64 + d
#pragma unroll
  for (int r = 0; r < 8; ++r) {
    float inv = 1.0f / lrow[r];
#pragma unroll
    for (int dt = 0; dt < 4; ++dt) {
      size_t row = (size_t)b * T_SEQ + i0 + r + 8 * hh;
      ctx[row * D_MODEL + h * HEAD_DIM + dt * 16 + n] =
          (_Float16)(acc[dt][r] * inv);
    }
  }
}

// ---- out projection + bias + alpha residual -> x1 (f32); 32x64 tile -------
__global__ __launch_bounds__(256) void k_oproj(const _Float16* __restrict__ ctx,
                                               const _Float16* __restrict__ Wh,
                                               const float* __restrict__ bout,
                                               const float* __restrict__ x,
                                               float* __restrict__ x1) {
  int lane = threadIdx.x & 31;
  int gw = blockIdx.x * 8 + (threadIdx.x >> 5);     // 2048 waves
  int ntg = gw % (D_MODEL / 64);
  int mt  = gw / (D_MODEL / 64);
  int m0 = mt * 32, n0 = ntg * 64;
  f8 acc[2][4];
#pragma unroll
  for (int mi = 0; mi < 2; ++mi)
#pragma unroll
    for (int nt = 0; nt < 4; ++nt) acc[mi][nt] = f8zero();
  for (int k0 = 0; k0 < D_MODEL; k0 += 32) {
    h16 a0 = ldA_f16(ctx, D_MODEL, m0, k0, lane);
    h16 a1 = ldA_f16(ctx, D_MODEL, m0 + 16, k0, lane);
#pragma unroll
    for (int nt = 0; nt < 4; ++nt) {
      h16 b = ldB_rowmaj(Wh, D_MODEL, n0 + 16 * nt, k0, lane);
      acc[0][nt] = wmma16(a0, b, acc[0][nt]);
      acc[1][nt] = wmma16(a1, b, acc[1][nt]);
    }
  }
  int n = lane & 15, hh = lane >> 4;
#pragma unroll
  for (int mi = 0; mi < 2; ++mi) {
#pragma unroll
    for (int nt = 0; nt < 4; ++nt) {
#pragma unroll
      for (int r = 0; r < 8; ++r) {
        size_t m = (size_t)m0 + 16 * mi + r + 8 * hh;
        int o = n0 + nt * 16 + n;
        x1[m * D_MODEL + o] =
            acc[mi][nt][r] + bout[o] + ALPHA_F * x[m * D_MODEL + o];
      }
    }
  }
}

// ---- RMSNorm: one wave per row --------------------------------------------
__global__ __launch_bounds__(256) void k_rms(const float* __restrict__ in,
                                             const float* __restrict__ g,
                                             float* __restrict__ out) {
  int lane = threadIdx.x & 31;
  int row  = blockIdx.x * 8 + (threadIdx.x >> 5);
  const float* p = in + (size_t)row * D_MODEL;
  float ss = 0.0f;
  for (int j = lane; j < D_MODEL; j += 32) {
    float v = p[j];
    ss += v * v;
  }
#pragma unroll
  for (int off = 1; off < 32; off <<= 1) ss += __shfl_xor(ss, off, 32);
  float rs = rsqrtf(ss / (float)D_MODEL + EPS_F);
  float* q = out + (size_t)row * D_MODEL;
  for (int j = lane; j < D_MODEL; j += 32) q[j] = p[j] * rs * g[j];
}

// ---- FFN1: yv & gate tiles in same wave, fused SwiGLU -> ff (f16) ---------
// 32 rows x 64 cols (x2 for gate) per wave
__global__ __launch_bounds__(256) void k_ffn1(const float* __restrict__ x2,
                                              const _Float16* __restrict__ W1h,
                                              _Float16* __restrict__ ff) {
  int lane = threadIdx.x & 31;
  int gw = blockIdx.x * 8 + (threadIdx.x >> 5);     // 8192 waves
  int ntg = gw % (H_FF / 64);                       // 32 groups
  int mt  = gw / (H_FF / 64);                       // 256 m-blocks
  int m0 = mt * 32, n0 = ntg * 64;
  f8 av[2][4], ag[2][4];
#pragma unroll
  for (int mi = 0; mi < 2; ++mi)
#pragma unroll
    for (int nt = 0; nt < 4; ++nt) { av[mi][nt] = f8zero(); ag[mi][nt] = f8zero(); }
  for (int k0 = 0; k0 < D_MODEL; k0 += 32) {
    h16 a0 = ldA_f32(x2, D_MODEL, m0, k0, lane);
    h16 a1 = ldA_f32(x2, D_MODEL, m0 + 16, k0, lane);
#pragma unroll
    for (int nt = 0; nt < 4; ++nt) {
      h16 bv = ldB_rowmaj(W1h, D_MODEL, n0 + 16 * nt, k0, lane);
      h16 bg = ldB_rowmaj(W1h, D_MODEL, H_FF + n0 + 16 * nt, k0, lane);
      av[0][nt] = wmma16(a0, bv, av[0][nt]);
      av[1][nt] = wmma16(a1, bv, av[1][nt]);
      ag[0][nt] = wmma16(a0, bg, ag[0][nt]);
      ag[1][nt] = wmma16(a1, bg, ag[1][nt]);
    }
  }
  int n = lane & 15, hh = lane >> 4;
#pragma unroll
  for (int mi = 0; mi < 2; ++mi) {
#pragma unroll
    for (int nt = 0; nt < 4; ++nt) {
#pragma unroll
      for (int r = 0; r < 8; ++r) {
        size_t m = (size_t)m0 + 16 * mi + r + 8 * hh;
        int o = n0 + nt * 16 + n;
        float yv = av[mi][nt][r], gt = ag[mi][nt][r];
        float silu = gt / (1.0f + __expf(-gt));
        ff[m * H_FF + o] = (_Float16)(yv * silu);
      }
    }
  }
}

// ---- FFN2: x3 = ff @ W2^T + alpha*x2 (f32); 32x64 tile --------------------
__global__ __launch_bounds__(256) void k_ffn2(const _Float16* __restrict__ ff,
                                              const _Float16* __restrict__ W2h,
                                              const float* __restrict__ x2,
                                              float* __restrict__ x3) {
  int lane = threadIdx.x & 31;
  int gw = blockIdx.x * 8 + (threadIdx.x >> 5);     // 2048 waves
  int ntg = gw % (D_MODEL / 64);
  int mt  = gw / (D_MODEL / 64);
  int m0 = mt * 32, n0 = ntg * 64;
  f8 acc[2][4];
#pragma unroll
  for (int mi = 0; mi < 2; ++mi)
#pragma unroll
    for (int nt = 0; nt < 4; ++nt) acc[mi][nt] = f8zero();
  for (int k0 = 0; k0 < H_FF; k0 += 32) {
    h16 a0 = ldA_f16(ff, H_FF, m0, k0, lane);
    h16 a1 = ldA_f16(ff, H_FF, m0 + 16, k0, lane);
#pragma unroll
    for (int nt = 0; nt < 4; ++nt) {
      h16 b = ldB_rowmaj(W2h, H_FF, n0 + 16 * nt, k0, lane);
      acc[0][nt] = wmma16(a0, b, acc[0][nt]);
      acc[1][nt] = wmma16(a1, b, acc[1][nt]);
    }
  }
  int n = lane & 15, hh = lane >> 4;
#pragma unroll
  for (int mi = 0; mi < 2; ++mi) {
#pragma unroll
    for (int nt = 0; nt < 4; ++nt) {
#pragma unroll
      for (int r = 0; r < 8; ++r) {
        size_t m = (size_t)m0 + 16 * mi + r + 8 * hh;
        int o = n0 + nt * 16 + n;
        x3[m * D_MODEL + o] = acc[mi][nt][r] + ALPHA_F * x2[m * D_MODEL + o];
      }
    }
  }
}

// ---------------------------------------------------------------------------
extern "C" void kernel_launch(void* const* d_in, const int* in_sizes, int n_in,
                              void* d_out, int out_size, void* d_ws,
                              size_t ws_size, hipStream_t stream) {
  (void)in_sizes; (void)n_in; (void)out_size; (void)ws_size;
  const float* x    = (const float*)d_in[0];
  const float* Wqkv = (const float*)d_in[1];
  const float* Wout = (const float*)d_in[2];
  const float* bout = (const float*)d_in[3];
  const float* W1   = (const float*)d_in[4];
  const float* W2   = (const float*)d_in[5];
  const float* g1   = (const float*)d_in[6];
  const float* g2   = (const float*)d_in[7];
  float* out = (float*)d_out;
  char* ws = (char*)d_ws;

  // arena layout (bytes); aliases: ff over [q..ctx], x3 over x1
  const size_t off_wqkv = 0;                         // 1536*512*2 = 1.5 MB
  const size_t off_wout = off_wqkv + (size_t)O_QKV * D_MODEL * 2;
  const size_t off_w1   = off_wout + (size_t)D_MODEL * D_MODEL * 2;
  const size_t off_w2   = off_w1 + (size_t)2 * H_FF * D_MODEL * 2;
  const size_t off_q    = off_w2 + (size_t)D_MODEL * H_FF * 2;
  const size_t sz_head  = (size_t)N_BATCH * NHEAD * T_SEQ * HEAD_DIM * 2;
  const size_t off_k    = off_q + sz_head;
  const size_t off_vT   = off_k + sz_head;
  const size_t off_ctx  = off_vT + sz_head;
  const size_t off_x1   = off_ctx + (size_t)M_ROWS * D_MODEL * 2;
  const size_t off_x2   = off_x1 + (size_t)M_ROWS * D_MODEL * 4;
  const size_t off_ff   = off_q;                     // alias (33.5 MB region)
  const size_t off_x3   = off_x1;                    // alias

  _Float16* wqkv_h = (_Float16*)(ws + off_wqkv);
  _Float16* wout_h = (_Float16*)(ws + off_wout);
  _Float16* w1_h   = (_Float16*)(ws + off_w1);
  _Float16* w2_h   = (_Float16*)(ws + off_w2);
  _Float16* q_h    = (_Float16*)(ws + off_q);
  _Float16* k_h    = (_Float16*)(ws + off_k);
  _Float16* vT_h   = (_Float16*)(ws + off_vT);
  _Float16* ctx_h  = (_Float16*)(ws + off_ctx);
  float*    x1     = (float*)(ws + off_x1);
  float*    x2     = (float*)(ws + off_x2);
  _Float16* ff_h   = (_Float16*)(ws + off_ff);
  float*    x3     = (float*)(ws + off_x3);

  // 1) weights f32 -> f16
  k_cvt<<<(O_QKV * D_MODEL + 255) / 256, 256, 0, stream>>>(Wqkv, wqkv_h,
                                                           O_QKV * D_MODEL);
  k_cvt<<<(D_MODEL * D_MODEL + 255) / 256, 256, 0, stream>>>(Wout, wout_h,
                                                             D_MODEL * D_MODEL);
  k_cvt<<<(2 * H_FF * D_MODEL + 255) / 256, 256, 0, stream>>>(
      W1, w1_h, 2 * H_FF * D_MODEL);
  k_cvt<<<(D_MODEL * H_FF + 255) / 256, 256, 0, stream>>>(W2, w2_h,
                                                          D_MODEL * H_FF);
  // 2) QKV GEMM (6144 waves, 32x64 tiles)
  k_qkv<<<(M_ROWS / 32) * (O_QKV / 64) / 8, 256, 0, stream>>>(x, wqkv_h, q_h,
                                                              k_h, vT_h);
  // 3) RoPE in place
  k_rope<<<(N_BATCH * NHEAD * T_SEQ * 32) / 256, 256, 0, stream>>>(q_h, k_h);
  // 4) banded flash attention (4096 waves)
  k_attn<<<(N_BATCH * NHEAD * (T_SEQ / 16)) / 8, 256, 0, stream>>>(q_h, k_h,
                                                                   vT_h, ctx_h);
  // 5) out projection + residual (2048 waves)
  k_oproj<<<(M_ROWS / 32) * (D_MODEL / 64) / 8, 256, 0, stream>>>(
      ctx_h, wout_h, bout, x, x1);
  // 6) RMSNorm -> x2
  k_rms<<<M_ROWS / 8, 256, 0, stream>>>(x1, g1, x2);
  // 7) FFN1 with fused SwiGLU (8192 waves)
  k_ffn1<<<(M_ROWS / 32) * (H_FF / 64) / 8, 256, 0, stream>>>(x2, w1_h, ff_h);
  // 8) FFN2 + residual (2048 waves)
  k_ffn2<<<(M_ROWS / 32) * (D_MODEL / 64) / 8, 256, 0, stream>>>(ff_h, w2_h, x2,
                                                                 x3);
  // 9) final RMSNorm -> out
  k_rms<<<M_ROWS / 8, 256, 0, stream>>>(x3, g2, out);
}
